// MultiHeadAttention_13262859010698
// MI455X (gfx1250) — matrix operand order repaired and checked by reading, not compiled, not run on hardware
//
#include <hip/hip_runtime.h>

// ---------------------------------------------------------------------------
// MI455X (gfx1250) multi-head attention forward — fused attention version.
// wave32, v_wmma_f32_16x16x32_bf16 for all matmuls, f32 accumulate.
// Scores tensor (268 MB) is written to HBM exactly once; softmax + P.V are
// fused on a 16x2048 f32 LDS tile (~132 KB of the WGP's 320 KB LDS).
// ---------------------------------------------------------------------------

typedef __attribute__((ext_vector_type(16))) __bf16 v16bf;
typedef __attribute__((ext_vector_type(8)))  float  v8f;

#define B_  2
#define S_  2048
#define D_  512
#define H_  8
#define HD_ 64
#define PPAD 2052   // 2048 + 4 floats row pad (LDS bank de-conflict)

__device__ __forceinline__ __bf16 f2bf(float x) { return (__bf16)x; }

__device__ __forceinline__ v8f wmma_bf16(v16bf a, v16bf b, v8f c) {
  // (neg_a, A, neg_b, B, c_mod, C, reuse_a, reuse_b)
  return __builtin_amdgcn_wmma_f32_16x16x32_bf16(false, a, false, b, (short)0, c,
                                                 false, false);
}

// A fragment: 16x32 bf16 row-major, row stride ld.
// lane L: M = L&15; half h = L>>4 -> K chunks {8h..8h+7, 16+8h..16+8h+7}.
__device__ __forceinline__ v16bf load_a_bf(const __bf16* __restrict__ base, int ld) {
  const int lane = threadIdx.x & 31;
  const int m = lane & 15, hh = lane >> 4;
  const __bf16* row = base + (size_t)m * ld;
  v16bf a;
#pragma unroll
  for (int i = 0; i < 8; ++i) {
    a[i]     = row[8 * hh + i];
    a[8 + i] = row[16 + 8 * hh + i];
  }
  return a;
}

// Same A layout, f32 source (LDS attention probs), convert on load.
__device__ __forceinline__ v16bf load_a_f32(const float* base, int ld) {
  const int lane = threadIdx.x & 31;
  const int m = lane & 15, hh = lane >> 4;
  const float* row = base + (size_t)m * ld;
  v16bf a;
#pragma unroll
  for (int i = 0; i < 8; ++i) {
    a[i]     = f2bf(row[8 * hh + i]);
    a[8 + i] = f2bf(row[16 + 8 * hh + i]);
  }
  return a;
}

// B fragment: 32x16, B[kk][n] = src[n*ld + kk] (src holds N rows of K dim).
// lane L: N = L&15; half h -> K chunks {8h..8h+7, 16+8h..16+8h+7}.
__device__ __forceinline__ v16bf load_b_bf(const __bf16* __restrict__ base, int ld) {
  const int lane = threadIdx.x & 31;
  const int n = lane & 15, hh = lane >> 4;
  const __bf16* row = base + (size_t)n * ld;
  v16bf b;
#pragma unroll
  for (int i = 0; i < 8; ++i) {
    b[i]     = row[8 * hh + i];
    b[8 + i] = row[16 + 8 * hh + i];
  }
  return b;
}

// ---------------------------------------------------------------------------
__global__ __launch_bounds__(256)
void cvt_f32_bf16(const float* __restrict__ src, __bf16* __restrict__ dst, int n) {
  int i = blockIdx.x * 256 + threadIdx.x;
  if (i < n) dst[i] = f2bf(src[i]);
}

// ---------------------------------------------------------------------------
// Projection GEMM: Y[16 x 512] = X_tile[16 x 512] @ W^T, then epilogue.
//   mode 0: RoPE + head split      -> out_bf [B,H,S,64]  (Q, K)
//   mode 1: head split + transpose -> out_bf [B,H,64,S]  (V)
//   mode 2: plain f32 store        -> out_f  [B*S, 512]  (final O-proj)
// ---------------------------------------------------------------------------
__global__ __launch_bounds__(256)
void proj_kernel(const __bf16* __restrict__ X, const __bf16* __restrict__ W,
                 __bf16* __restrict__ out_bf, float* __restrict__ out_f, int mode) {
  __shared__ float Y[16][512];          // 32 KB
  const int mtile = blockIdx.x;         // 0..255
  const int tid  = threadIdx.x;
  const int wave = tid >> 5;
  const int lane = tid & 31;
  const int l15  = lane & 15;
  const int hl   = lane >> 4;

  v8f acc[4];
#pragma unroll
  for (int t = 0; t < 4; ++t)
#pragma unroll
    for (int r = 0; r < 8; ++r) acc[t][r] = 0.0f;

  const __bf16* Xbase = X + (size_t)mtile * 16 * D_;
  for (int kk = 0; kk < D_; kk += 32) {
    v16bf a = load_a_bf(Xbase + kk, D_);
    if (kk + 32 < D_)  // stream next W stripe through L2 (global_prefetch_b8)
      __builtin_prefetch(W + (size_t)(wave * 64 + l15 * 4) * D_ + kk + 32, 0, 0);
#pragma unroll
    for (int t = 0; t < 4; ++t) {
      const __bf16* bptr = W + (size_t)(wave * 64 + t * 16) * D_ + kk;
      acc[t] = wmma_bf16(a, load_b_bf(bptr, D_), acc[t]);
    }
  }

#pragma unroll
  for (int t = 0; t < 4; ++t) {
    const int n0 = wave * 64 + t * 16 + l15;
#pragma unroll
    for (int r = 0; r < 8; ++r) Y[r + 8 * hl][n0] = acc[t][r];
  }
  __syncthreads();

  const int mglob0 = mtile * 16;
  if (mode == 0) {
    // RoPE over the full 512-dim row, then head split.
    for (int p = tid; p < 16 * 256; p += 256) {
      const int r = p >> 8, i = p & 255;
      const int mg = mglob0 + r;
      const int b = mg >> 11, s = mg & 2047;
      const float fr = __expf((float)i * -0.035977896f);  // -ln(10000)/256
      float sn, cs;
      __sincosf((float)s * fr, &sn, &cs);
      const float e = Y[r][2 * i], o = Y[r][2 * i + 1];
      const float v0 = e * cs - o * sn;
      const float v1 = o * cs + e * sn;
      const int c0 = i, c1 = 256 + i;
      out_bf[(((size_t)(b * H_ + (c0 >> 6)) * S_ + s) * HD_) + (c0 & 63)] = f2bf(v0);
      out_bf[(((size_t)(b * H_ + (c1 >> 6)) * S_ + s) * HD_) + (c1 & 63)] = f2bf(v1);
    }
  } else if (mode == 1) {
    for (int p = tid; p < 16 * 512; p += 256) {
      const int r = p >> 9, c = p & 511;
      const int mg = mglob0 + r;
      const int b = mg >> 11, s = mg & 2047;
      out_bf[((size_t)(b * H_ + (c >> 6)) * HD_ + (c & 63)) * S_ + s] = f2bf(Y[r][c]);
    }
  } else {
    for (int p = tid; p < 16 * 512; p += 256) {
      const int r = p >> 9, c = p & 511;
      out_f[(size_t)(mglob0 + r) * D_ + c] = Y[r][c];
    }
  }
}

// ---------------------------------------------------------------------------
// Fused attention: one workgroup = (b, h, 16 query rows).
//  P1: logits 16x2048 -> LDS (f32, scaled + masked)
//  P2: per-row softmax, one row per 16-lane half-wave; LDS keeps exp(v-max),
//      inv[] keeps 1/sum (normalization folded into consumers)
//  P3: coalesced float4 write of normalized scores to d_out
//  P4: E = P.V via WMMA (P from LDS f32->bf16, V bf16 transposed [B,H,64,S]);
//      2 deterministic ds_add_f32 contributions per element; scale by inv[r].
// LDS: 16*2052*4 + 16*64*4 + 64 = ~132 KB  (2 workgroups / 320 KB WGP)
// ---------------------------------------------------------------------------
__global__ __launch_bounds__(256)
void attn_fused_kernel(const __bf16* __restrict__ Q, const __bf16* __restrict__ K,
                       const __bf16* __restrict__ Vt, const float* __restrict__ mask,
                       float* __restrict__ scores, __bf16* __restrict__ E) {
  __shared__ float P[16][PPAD];
  __shared__ float Es[16][64];
  __shared__ float inv_l[16];

  const int blk   = blockIdx.x;        // B*H*(S/16) = 2048
  const int qtile = blk & 127;
  const int bh    = blk >> 7;          // 0..15
  const int b     = bh >> 3, h = bh & 7;
  const int tid  = threadIdx.x;
  const int wave = tid >> 5;
  const int lane = tid & 31;
  const int l15  = lane & 15;
  const int hl   = lane >> 4;
  const int q0   = qtile * 16;

  for (int p = tid; p < 16 * 64; p += 256) (&Es[0][0])[p] = 0.0f;

  // ---- Phase 1: logits -> LDS ----
  const __bf16* Qbase = Q + ((size_t)bh * S_ + q0) * HD_;
  const v16bf a0 = load_a_bf(Qbase + 0, HD_);
  const v16bf a1 = load_a_bf(Qbase + 32, HD_);

  for (int t = 0; t < 16; ++t) {
    const int nt = wave * 256 + t * 16;
    const __bf16* kb = K + ((size_t)bh * S_ + nt) * HD_;
    if (t + 1 < 16)  // prefetch next K tile's row for this lane
      __builtin_prefetch(kb + (size_t)(16 + l15) * HD_, 0, 0);
    v8f acc;
#pragma unroll
    for (int r = 0; r < 8; ++r) acc[r] = 0.0f;
    acc = wmma_bf16(a0, load_b_bf(kb, HD_), acc);
    acc = wmma_bf16(a1, load_b_bf(kb + 32, HD_), acc);
    const int n = nt + l15;
#pragma unroll
    for (int r = 0; r < 8; ++r) {
      const int row = r + 8 * hl;
      const float mval = mask[((size_t)b * S_ + q0 + row) * S_ + n];
      P[row][n] = (acc[r] + mval * -1.0e9f) * 0.125f;
    }
  }
  __syncthreads();

  // ---- Phase 2: softmax, one row per 16-lane half ----
  {
    const int row = 2 * wave + hl;
    const int j = l15;
    float m = -3.0e38f;
#pragma unroll 4
    for (int i = 0; i < 128; ++i) m = fmaxf(m, P[row][j + 16 * i]);
#pragma unroll
    for (int o = 8; o > 0; o >>= 1) m = fmaxf(m, __shfl_xor(m, o, 32));
    float s = 0.0f;
#pragma unroll 4
    for (int i = 0; i < 128; ++i) {
      const int c = j + 16 * i;
      const float e = __expf(P[row][c] - m);
      P[row][c] = e;
      s += e;
    }
#pragma unroll
    for (int o = 8; o > 0; o >>= 1) s += __shfl_xor(s, o, 32);
    if (j == 0) inv_l[row] = 1.0f / s;
  }
  __syncthreads();

  // ---- Phase 3: normalized scores -> global (float4, coalesced) ----
  {
    const int r = tid >> 4, j = tid & 15;
    const float iv = inv_l[r];
    float* dst = scores + ((size_t)bh * S_ + q0 + r) * S_;
#pragma unroll 4
    for (int it = 0; it < 32; ++it) {
      const int c = j * 4 + it * 64;
      float4 v;
      v.x = P[r][c + 0] * iv;
      v.y = P[r][c + 1] * iv;
      v.z = P[r][c + 2] * iv;
      v.w = P[r][c + 3] * iv;
      *(float4*)(dst + c) = v;
    }
  }

  // ---- Phase 4: E = P.V ----
  {
    const int ntile = wave & 3, khalf = wave >> 2;
    const __bf16* Vbase = Vt + ((size_t)bh * HD_ + ntile * 16) * S_;
    v8f acc;
#pragma unroll
    for (int r = 0; r < 8; ++r) acc[r] = 0.0f;
    for (int step = 0; step < 32; ++step) {
      const int kbase = khalf * 1024 + step * 32;
      if (step + 1 < 32)  // prefetch next V stripe for this lane
        __builtin_prefetch(Vbase + (size_t)l15 * S_ + kbase + 32, 0, 0);
      v16bf a = load_a_f32(&P[0][0] + kbase, PPAD);
      v16bf bf = load_b_bf(Vbase + kbase, S_);
      acc = wmma_bf16(a, bf, acc);
    }
#pragma unroll
    for (int r = 0; r < 8; ++r)
      atomicAdd(&Es[r + 8 * hl][ntile * 16 + l15], acc[r]);
  }
  __syncthreads();

  for (int p = tid; p < 16 * 64; p += 256) {
    const int r = p >> 6, c = p & 63;
    const int s = q0 + r;
    E[((size_t)(b * S_ + s)) * D_ + h * HD_ + c] = f2bf(Es[r][c] * inv_l[r]);
  }
}

// ---------------------------------------------------------------------------
extern "C" void kernel_launch(void* const* d_in, const int* in_sizes, int n_in,
                              void* d_out, int out_size, void* d_ws, size_t ws_size,
                              hipStream_t stream) {
  const float* querys = (const float*)d_in[0];
  const float* keys   = (const float*)d_in[1];
  const float* values = (const float*)d_in[2];
  const float* mask   = (const float*)d_in[3];
  const float* Wq     = (const float*)d_in[4];
  const float* Wk     = (const float*)d_in[5];
  const float* Wv     = (const float*)d_in[6];
  const float* Wo     = (const float*)d_in[7];

  float* out_emb    = (float*)d_out;                    // [B,S,512]
  float* out_scores = out_emb + (size_t)B_ * S_ * D_;   // [B,H,S,S]

  char* ws = (char*)d_ws;
  size_t off = 0;
  auto alloc = [&](size_t bytes) -> void* {
    void* p = (void*)(ws + off);
    off += (bytes + 255) & ~size_t(255);
    return p;
  };
  const size_t NX = (size_t)B_ * S_ * D_;   // 2,097,152
  const size_t NW = (size_t)D_ * D_;        // 262,144
  __bf16* Xq  = (__bf16*)alloc(NX * 2);
  __bf16* Xk  = (__bf16*)alloc(NX * 2);
  __bf16* Xv  = (__bf16*)alloc(NX * 2);
  __bf16* Wqb = (__bf16*)alloc(NW * 2);
  __bf16* Wkb = (__bf16*)alloc(NW * 2);
  __bf16* Wvb = (__bf16*)alloc(NW * 2);
  __bf16* Wob = (__bf16*)alloc(NW * 2);
  __bf16* Qbf = (__bf16*)alloc(NX * 2);     // [B,H,S,64]
  __bf16* Kbf = (__bf16*)alloc(NX * 2);     // [B,H,S,64]
  __bf16* Vtb = (__bf16*)alloc(NX * 2);     // [B,H,64,S]
  __bf16* Ebf = (__bf16*)alloc(NX * 2);     // [B*S, 512]
  (void)ws_size; (void)in_sizes; (void)n_in; (void)out_size;

  const int TB = 256;
  cvt_f32_bf16<<<(int)((NX + TB - 1) / TB), TB, 0, stream>>>(querys, Xq, (int)NX);
  cvt_f32_bf16<<<(int)((NX + TB - 1) / TB), TB, 0, stream>>>(keys,   Xk, (int)NX);
  cvt_f32_bf16<<<(int)((NX + TB - 1) / TB), TB, 0, stream>>>(values, Xv, (int)NX);
  cvt_f32_bf16<<<(int)((NW + TB - 1) / TB), TB, 0, stream>>>(Wq, Wqb, (int)NW);
  cvt_f32_bf16<<<(int)((NW + TB - 1) / TB), TB, 0, stream>>>(Wk, Wkb, (int)NW);
  cvt_f32_bf16<<<(int)((NW + TB - 1) / TB), TB, 0, stream>>>(Wv, Wvb, (int)NW);
  cvt_f32_bf16<<<(int)((NW + TB - 1) / TB), TB, 0, stream>>>(Wo, Wob, (int)NW);

  // Q/K/V projections (+RoPE for Q,K; transpose for V)
  proj_kernel<<<(B_ * S_) / 16, TB, 0, stream>>>(Xq, Wqb, Qbf, nullptr, 0);
  proj_kernel<<<(B_ * S_) / 16, TB, 0, stream>>>(Xk, Wkb, Kbf, nullptr, 0);
  proj_kernel<<<(B_ * S_) / 16, TB, 0, stream>>>(Xv, Wvb, Vtb, nullptr, 1);

  // Fused logits + softmax + scores-write + P.V (scores hit HBM exactly once)
  attn_fused_kernel<<<B_ * H_ * (S_ / 16), TB, 0, stream>>>(Qbf, Kbf, Vtb, mask,
                                                            out_scores, Ebf);

  // Output projection -> f32 embeds
  proj_kernel<<<(B_ * S_) / 16, TB, 0, stream>>>(Ebf, Wob, nullptr, out_emb, 2);
}
